// WaveletTransform3Channel_27058293965517
// MI455X (gfx1250) — compile-verified, matrix operand order checked by reading, not run on hardware
//
#include <hip/hip_runtime.h>
#include <hip/hip_bf16.h>

// CDNA5 / gfx1250: wave32, WMMA 16x16x32 bf16 (codegen-confirmed builtin).
typedef __attribute__((ext_vector_type(16))) __bf16 v16bf;
typedef __attribute__((ext_vector_type(8)))  float  v8f;

#define NBAND 3
#define MT    16    // freq rows per tile (WMMA M)
#define NTILE 128   // time outputs per block (8 waves x 16)
#define KC    128   // K chunk staged in LDS (4 WMMA K-steps)
#define AS    136   // LDS row stride (bf16 elems) for A tiles: 272B -> bank-staggered rows

union BFrag { unsigned int u[8]; v16bf v; };

// ---------------------------------------------------------------------------
// Kernel 0: zero the output (we accumulate with atomics) + init tile K-ranges.
// ---------------------------------------------------------------------------
__global__ void wt_init_kernel(int* __restrict__ ranges, int nMT,
                               float* __restrict__ out, int outN) {
  int tid = blockIdx.x * blockDim.x + threadIdx.x;
  if (blockIdx.x == 0 && (int)threadIdx.x < nMT * 2) {
    ranges[threadIdx.x] = (threadIdx.x & 1) ? -1 : 0x7FFFFFFF; // [min,max] per tile
  }
  for (int i = tid; i < outN; i += gridDim.x * blockDim.x) out[i] = 0.0f;
}

// ---------------------------------------------------------------------------
// Kernel 1: build wd[f,m] = nss[f]*(w[f,m-1]-w[f,m])  (K = Lmax+1, folded diff),
// split into bf16 hi/lo planes, zero-pad rows to nMT*16 and K to Kpad, and
// record per-16-row-tile nonzero support [kmin,kmax] (data-driven K skipping).
// ---------------------------------------------------------------------------
__global__ void wt_prep_weights(const float* __restrict__ W,
                                const float* __restrict__ nss,
                                __bf16* __restrict__ gAhi,
                                __bf16* __restrict__ gAlo,
                                int* __restrict__ ranges,
                                int nF, int Lmax, int K, int Kpad) {
  const int f = blockIdx.x; // 0 .. nMT*16-1 (rows >= nF are zero pad)
  __shared__ int smin, smax;
  if (threadIdx.x == 0) { smin = 0x7FFFFFFF; smax = -1; }
  __syncthreads();

  int lmin = 0x7FFFFFFF, lmax = -1;
  const float scale = (f < nF) ? nss[f] : 0.0f;
  const float* Wr = W + (size_t)f * Lmax;
  for (int m = threadIdx.x; m < Kpad; m += blockDim.x) {
    float wd = 0.0f;
    if (f < nF && m < K) {
      float wm1 = (m >= 1)   ? Wr[m - 1] : 0.0f;
      float wm  = (m < Lmax) ? Wr[m]     : 0.0f;
      wd = scale * (wm1 - wm);
    }
    __bf16 h = (__bf16)wd;
    __bf16 l = (__bf16)(wd - (float)h);
    gAhi[(size_t)f * Kpad + m] = h;
    gAlo[(size_t)f * Kpad + m] = l;
    if (wd != 0.0f) { lmin = min(lmin, m); lmax = max(lmax, m); }
  }
  atomicMin(&smin, lmin);
  atomicMax(&smax, lmax);
  __syncthreads();
  if (threadIdx.x == 0 && f < nF && smax >= 0) {
    atomicMin(&ranges[(f >> 4) * 2 + 0], smin);
    atomicMax(&ranges[(f >> 4) * 2 + 1], smax);
  }
}

// ---------------------------------------------------------------------------
// Kernel 2: Toeplitz GEMM via v_wmma_f32_16x16x32_bf16 with hi/lo compensation
// (3 WMMAs per K-step ~ fp32 accuracy at bf16 matrix rates), then |.| and
// masked band accumulation into out (8,3,64,nT).
// Block = (signal, 16-freq tile, 128 time outputs), 8 waves of 32.
// x window is pre-split into packed {lo_bf16,hi_bf16} uints at staging time so
// the MAC loop does only LDS gathers + cheap bitfield merges (no cvt chains).
// ---------------------------------------------------------------------------
__global__ __launch_bounds__(256)
void wt_band_gemm(const float* __restrict__ x,
                  const float* __restrict__ masks,
                  const float* __restrict__ counts,
                  const __bf16* __restrict__ gAhi,
                  const __bf16* __restrict__ gAlo,
                  const int* __restrict__ ranges,
                  float* __restrict__ out,
                  int nF, int nT, int Kpad, int padL) {
  const int ttile = blockIdx.x;
  const int mtile = blockIdx.y;
  const int sig   = blockIdx.z;
  const int tid   = threadIdx.x;
  const int lane  = tid & 31;
  const int wave  = tid >> 5;
  const int half  = lane >> 4;   // wave32: lane halves hold different K sets
  const int mn    = lane & 15;   // A: row m   B: column n

  __shared__ __bf16       sAhi[MT * AS];
  __shared__ __bf16       sAlo[MT * AS];
  __shared__ unsigned int sXp[2 * NTILE];  // packed {lo,hi} bf16 per x element
  __shared__ float        sBC[NBAND][MT];  // mask[k,f]/count[k] for this tile
  __shared__ int          sAny;

  if (tid == 0) sAny = 0;
  __syncthreads();
  if (tid < NBAND * MT) {
    int k = tid >> 4, fl = tid & 15;
    int f = mtile * MT + fl;
    float m = (f < nF) ? masks[k * nF + f] : 0.0f;
    float v = m / counts[k];
    sBC[k][fl] = v;
    if (v != 0.0f) atomicOr(&sAny, 1);
  }
  __syncthreads();
  if (sAny == 0) return;  // this freq tile contributes to no band: skip all work

  const int kmin = ranges[mtile * 2 + 0];
  const int kmax = ranges[mtile * 2 + 1];

  v8f c = {};
  if (kmin <= kmax) {
    const int t0 = ttile * NTILE;
    const size_t xbase = (size_t)sig * nT;
    const int kStart = kmin & ~31;

    for (int kb = kStart; kb <= kmax; kb += KC) {
      __syncthreads(); // protect previous iteration's fragment reads
      // Stage x window pre-split: sXp[j] = pack(xpad[t0 + kb + j]).
      // One cvt pair per element here vs ~64 redundant splits in the MAC loop.
      {
        int j   = tid;                 // 0..255
        int idx = t0 + kb + j - padL;  // padL = Lmax-1-S left pad
        float v = (idx >= 0 && idx < nT) ? x[xbase + idx] : 0.0f;
        __bf16 h = (__bf16)v;
        __bf16 l = (__bf16)(v - (float)h);
        sXp[j] = (unsigned int)__builtin_bit_cast(unsigned short, h) |
                 ((unsigned int)__builtin_bit_cast(unsigned short, l) << 16);
      }
      // Stage weight tile hi/lo: 16 rows x 128 K (coalesced bf16 loads).
      for (int i = tid; i < MT * KC; i += 256) {
        int r  = i >> 7;
        int cc = i & (KC - 1);
        int kk = kb + cc;
        bool ok = kk < Kpad;
        size_t g = (size_t)(mtile * MT + r) * Kpad + kk;
        sAhi[r * AS + cc] = ok ? gAhi[g] : (__bf16)0.0f;
        sAlo[r * AS + cc] = ok ? gAlo[g] : (__bf16)0.0f;
      }
      // Prefetch next chunk of the weight planes (global_prefetch_b8).
      if (kb + KC <= kmax) {
        __builtin_prefetch(&gAhi[(size_t)(mtile * MT + (tid & 15)) * Kpad + kb + KC], 0, 1);
      }
      __syncthreads();

      // 4 K-steps of 32; same (lane-half,slot)->k map for A and B fragments,
      // so the contraction is exact for any consistent hardware K ordering.
      #pragma unroll
      for (int s32 = 0; s32 < KC; s32 += 32) {
        v16bf ahi, alo;
        BFrag bhi, blo;
        const int jb = wave * 16 + mn + s32 + (half << 3);
        #pragma unroll
        for (int e = 0; e < 16; e += 2) {
          // ISA 16-bit A layout: half 0 slots -> K {0..7,16..23}, half 1 -> {8..15,24..31}
          int kk = (e & 7) + ((e >> 3) << 4);          // pair-contiguous
          unsigned int w0 = sXp[jb + kk];
          unsigned int w1 = sXp[jb + kk + 1];
          bhi.u[e >> 1] = (w0 & 0xFFFFu) | (w1 << 16);
          blo.u[e >> 1] = (w0 >> 16) | (w1 & 0xFFFF0000u);
        }
        #pragma unroll
        for (int e = 0; e < 16; ++e) {
          int kk = (e & 7) + (half << 3) + ((e >> 3) << 4);
          ahi[e] = sAhi[mn * AS + s32 + kk];
          alo[e] = sAlo[mn * AS + s32 + kk];
        }
        // D = A*B + C, f32 accumulate; hi*hi + hi*lo + lo*hi compensation.
        c = __builtin_amdgcn_wmma_f32_16x16x32_bf16(false, ahi, false, bhi.v, (short)0, c, false, false);
        c = __builtin_amdgcn_wmma_f32_16x16x32_bf16(false, ahi, false, blo.v, (short)0, c, false, false);
        c = __builtin_amdgcn_wmma_f32_16x16x32_bf16(false, alo, false, bhi.v, (short)0, c, false, false);
      }
    }
  }

  // Epilogue: C layout (16x16 f32): lane (mn, half), VGPR r -> (M = r + 8*half, N = mn).
  const int t = ttile * NTILE + wave * 16 + mn;
  if (t < nT) {
    const int b  = sig >> 6;
    const int ch = sig & 63;
    #pragma unroll
    for (int r = 0; r < 8; ++r) {
      float p  = fabsf(c[r]);
      int   fl = r + (half << 3);
      #pragma unroll
      for (int k = 0; k < NBAND; ++k) {
        float w = sBC[k][fl];
        if (w != 0.0f) {
          atomicAdd(&out[(((size_t)b * NBAND + k) * 64 + ch) * nT + t], w * p);
        }
      }
    }
  }
}

// ---------------------------------------------------------------------------
// Host launcher. Inputs: x(8,64,1000) f32, weights(150,1,Lmax) f32,
// neg_sqrt_scales(150) f32, band_masks(3,150) f32, band_counts(3) f32.
// Output: (8,3,64,1000) f32. Workspace: ~4.3 MB for bf16 hi/lo weight planes.
// ---------------------------------------------------------------------------
extern "C" void kernel_launch(void* const* d_in, const int* in_sizes, int n_in,
                              void* d_out, int out_size, void* d_ws, size_t ws_size,
                              hipStream_t stream) {
  const float* x      = (const float*)d_in[0];
  const float* W      = (const float*)d_in[1];
  const float* nss    = (const float*)d_in[2];
  const float* masks  = (const float*)d_in[3];
  const float* counts = (const float*)d_in[4];
  float* out          = (float*)d_out;

  const int nF   = in_sizes[2];            // 150
  const int Lmax = in_sizes[1] / nF;       // ~6501
  const int nT   = in_sizes[0] / (8 * 64); // 1000
  const int K    = Lmax + 1;               // diff folded into kernel
  const int Kpad = ((K + 127) / 128) * 128;
  const int S    = (Lmax - 2) / 2;
  const int padL = Lmax - 1 - S;
  const int nMT  = (nF + MT - 1) / MT;     // 10 freq tiles

  // Workspace layout: [ranges: nMT*2 ints][Ahi plane][Alo plane], bf16 planes
  // are (nMT*16) x Kpad row-major.
  char* ws = (char*)d_ws;
  int*    d_ranges = (int*)ws;
  size_t  off      = 256;
  __bf16* d_Ahi    = (__bf16*)(ws + off);
  off += (size_t)nMT * MT * Kpad * sizeof(__bf16);
  off  = (off + 255) & ~(size_t)255;
  __bf16* d_Alo    = (__bf16*)(ws + off);
  (void)ws_size; (void)n_in;

  wt_init_kernel<<<512, 256, 0, stream>>>(d_ranges, nMT, out, out_size);
  wt_prep_weights<<<nMT * MT, 256, 0, stream>>>(W, nss, d_Ahi, d_Alo, d_ranges,
                                                nF, Lmax, K, Kpad);
  dim3 grid((nT + NTILE - 1) / NTILE, nMT, 512);
  wt_band_gemm<<<grid, 256, 0, stream>>>(x, masks, counts, d_Ahi, d_Alo,
                                         d_ranges, out, nF, nT, Kpad, padL);
}
